// WindowAttention_82222853915234
// MI455X (gfx1250) — compile-verified, matrix operand order checked by reading, not run on hardware
//
#include <hip/hip_runtime.h>
#include <hip/hip_bf16.h>

typedef __attribute__((ext_vector_type(16))) __bf16 v16bf;
typedef __attribute__((ext_vector_type(8)))  float  v8f;

#define WMMA_BF16(a, b, c) \
  __builtin_amdgcn_wmma_f32_16x16x32_bf16(false, (a), false, (b), (short)0, (c), false, false)

#define HEADS 12
#define NSEQ  49

// --- optional gfx1250 async global->LDS path (guarded: falls back cleanly) ---
#if defined(__has_builtin)
#if __has_builtin(__builtin_amdgcn_global_load_async_to_lds_b128) && \
    __has_builtin(__builtin_amdgcn_s_wait_asynccnt)
#define USE_ASYNC_LDS 1
#endif
#endif

#ifdef USE_ASYNC_LDS
typedef int v4i_async __attribute__((vector_size(16)));
typedef v4i_async __attribute__((address_space(1))) * gas_v4i;
typedef v4i_async __attribute__((address_space(3))) * las_v4i;
static __device__ __forceinline__ void async_copy_b128(const void* g, void* l) {
  __builtin_amdgcn_global_load_async_to_lds_b128((gas_v4i)g, (las_v4i)l, 0, 0);
}
#endif

// Load a 16x32 bf16 fragment from a row-major fp32 source (global or LDS).
// ISA layout (CDNA5 ISA, 16-bit A 16x32): lane<16 -> K in {0..7}U{16..23},
// lane>=16 -> K in {8..15}U{24..31}: two contiguous runs of 8.
static __device__ __forceinline__ v16bf frag_from_f32(const float* __restrict__ src, int ld) {
  const int lane = threadIdx.x & 31;
  const int half = lane >> 4;
  const int row  = lane & 15;
  const float* p = src + (size_t)row * ld + half * 8;
  v16bf f;
#pragma unroll
  for (int e = 0; e < 8; ++e) f[e] = (__bf16)p[e];
#pragma unroll
  for (int e = 0; e < 8; ++e) f[8 + e] = (__bf16)p[16 + e];
  return f;
}

// Same fragment pattern from an LDS-resident bf16 tile.
static __device__ __forceinline__ v16bf frag_from_bf16(const __bf16* src, int ld) {
  const int lane = threadIdx.x & 31;
  const int half = lane >> 4;
  const int row  = lane & 15;
  const __bf16* p = src + row * ld + half * 8;
  v16bf f;
#pragma unroll
  for (int e = 0; e < 8; ++e) f[e] = p[e];
#pragma unroll
  for (int e = 0; e < 8; ++e) f[8 + e] = p[16 + e];
  return f;
}

// C[m,n] = sum_k A[m,k] * W[n,k] + bias[n] (+ addTo[m,n])
// One wave computes a 32(M) x 64(N) tile: 8 WMMAs per k-step, each B fragment
// reused by two A fragments. M%32==0, N%64==0, K%32==0 for all calls here.
__global__ __launch_bounds__(256)
void gemm_bf16_kernel(const float* __restrict__ A, const float* __restrict__ W,
                      const float* __restrict__ bias, const float* __restrict__ addTo,
                      float* __restrict__ C, int M, int N, int K) {
  const int wave    = blockIdx.x * (blockDim.x >> 5) + (threadIdx.x >> 5);
  const int nStrips = N >> 6;
  const int mTile   = wave / nStrips;
  const int nStrip  = wave - mTile * nStrips;
  const int m0      = mTile << 5;
  if (m0 >= M) return;  // uniform per wave: EXEC stays full for WMMA
  const int n0 = nStrip << 6;
  const int lane = threadIdx.x & 31;
  const int half = lane >> 4;
  const int lr   = lane & 15;

  v8f acc0[4] = {};
  v8f acc1[4] = {};
  for (int k0 = 0; k0 < K; k0 += 32) {
    if (k0 + 32 < K) {  // prefetch next k-step of the HBM-streaming operand
      __builtin_prefetch(A + (size_t)(m0 + lr) * K + k0 + 32 + half * 8, 0, 3);
      __builtin_prefetch(A + (size_t)(m0 + 16 + lr) * K + k0 + 32 + half * 8, 0, 3);
    }
    v16bf a0 = frag_from_f32(A + (size_t)m0 * K + k0, K);
    v16bf a1 = frag_from_f32(A + (size_t)(m0 + 16) * K + k0, K);
#pragma unroll
    for (int t = 0; t < 4; ++t) {
      v16bf b = frag_from_f32(W + (size_t)(n0 + 16 * t) * K + k0, K);
      acc0[t] = WMMA_BF16(a0, b, acc0[t]);
      acc1[t] = WMMA_BF16(a1, b, acc1[t]);
    }
  }

#pragma unroll
  for (int t = 0; t < 4; ++t) {
    const int n  = n0 + 16 * t + lr;
    const float bn = bias[n];
#pragma unroll
    for (int j = 0; j < 8; ++j) {
      const int m = m0 + j + 8 * half;
      float v0 = acc0[t][j] + bn;
      float v1 = acc1[t][j] + bn;
      if (addTo) {
        v0 += addTo[(size_t)m * N + n];
        v1 += addTo[(size_t)(m + 16) * N + n];
      }
      C[(size_t)m * N + n] = v0;
      C[(size_t)(m + 16) * N + n] = v1;
    }
  }
}

// Fused windowed self-attention for one (window b, head h) per block.
// qkv layout: (B_, 49, 3*384) with column = s*384 + h*32 + d.
__global__ __launch_bounds__(128)
void win_attn_kernel(const float* __restrict__ qkv, const float* __restrict__ rel_table,
                     const float* __restrict__ mask, float* __restrict__ attout) {
  const int b   = blockIdx.x / HEADS;
  const int h   = blockIdx.x % HEADS;
  const int tid = threadIdx.x;
  const int wv = tid >> 5, lane = tid & 31, half = lane >> 4, lr = lane & 15;

  __shared__ float  qf[64 * 32];   // q rows, fp32 (padded to 64)
  __shared__ float  kf[64 * 32];   // k rows, fp32
  __shared__ __bf16 vts[32 * 64];  // v transposed: [d][m]
  __shared__ __bf16 ps[64 * 64];   // softmax probabilities (bf16)
  __shared__ float  sc[64 * 64];   // raw scores

  const float scale = 0.1767766952966369f;  // 32^-0.5

  // zero the pad rows (49..63) of q/k with plain DS stores (disjoint from copies)
  for (int i = tid + NSEQ * 32; i < 64 * 32; i += 128) {
    qf[i] = 0.f;
    kf[i] = 0.f;
  }
#ifdef USE_ASYNC_LDS
  for (int c = tid; c < NSEQ * 8; c += 128) {  // 8 x 16B chunks per 32-float row
    const int n = c >> 3, f4 = (c & 7) << 2;
    const float* gq = qkv + ((size_t)b * NSEQ + n) * 1152 + h * 32 + f4;
    async_copy_b128(gq,       qf + n * 32 + f4);
    async_copy_b128(gq + 384, kf + n * 32 + f4);
  }
#else
  for (int i = tid; i < NSEQ * 32; i += 128) {
    const int n = i >> 5, d = i & 31;
    const float* p = qkv + ((size_t)b * NSEQ + n) * 1152 + h * 32 + d;
    qf[i] = p[0];
    kf[i] = p[384];
  }
#endif
  for (int i = tid; i < 32 * 64; i += 128) {
    const int d = i >> 6, m = i & 63;
    float vv = 0.f;
    if (m < NSEQ) vv = qkv[((size_t)b * NSEQ + m) * 1152 + 768 + h * 32 + d];
    vts[i] = (__bf16)vv;
  }
#ifdef USE_ASYNC_LDS
  __builtin_amdgcn_s_wait_asynccnt(0);  // drain this wave's async copies
#endif
  __syncthreads();

  // scores = q k^T : wave wv owns query rows [16wv, 16wv+16)
  {
    v16bf a = frag_from_f32(qf + (wv * 16) * 32, 32);
#pragma unroll
    for (int t = 0; t < 4; ++t) {
      v16bf kb = frag_from_f32(kf + (t * 16) * 32, 32);
      v8f s = {};
      s = WMMA_BF16(a, kb, s);
#pragma unroll
      for (int j = 0; j < 8; ++j) {
        const int row = wv * 16 + j + 8 * half;  // query index
        const int col = t * 16 + lr;             // key index
        float val = -1e30f;                      // mask padded key columns
        if (col < NSEQ) {
          val = s[j] * scale;
          if (row < NSEQ) {
            const int i1 = row / 7, j1 = row % 7, i2 = col / 7, j2 = col % 7;
            const int ridx = (i1 - i2 + 6) * 13 + (j1 - j2 + 6);
            val += rel_table[ridx * HEADS + h];
            val += mask[((size_t)(b & 63) * NSEQ + row) * NSEQ + col];
          }
        }
        sc[row * 64 + col] = val;
      }
    }
  }
  __syncthreads();

  // softmax over valid keys (one row per thread)
  if (tid < 64) {
    const int row = tid;
    if (row < NSEQ) {
      float mx = -1e30f;
      for (int m = 0; m < NSEQ; ++m) mx = fmaxf(mx, sc[row * 64 + m]);
      float sum = 0.f;
      for (int m = 0; m < NSEQ; ++m) {
        const float e = __expf(sc[row * 64 + m] - mx);
        sc[row * 64 + m] = e;
        sum += e;
      }
      const float inv = 1.f / sum;
      for (int m = 0; m < 64; ++m)
        ps[row * 64 + m] = (m < NSEQ) ? (__bf16)(sc[row * 64 + m] * inv) : (__bf16)0.f;
    } else {
      for (int m = 0; m < 64; ++m) ps[row * 64 + m] = (__bf16)0.f;
    }
  }
  __syncthreads();

  // out = P V : K=64 (2 k-steps), 2 N-tiles of head_dim 32
  v8f oacc[2] = {};
  for (int k0 = 0; k0 < 64; k0 += 32) {
    v16bf ap = frag_from_bf16(ps + (wv * 16) * 64 + k0, 64);
#pragma unroll
    for (int t = 0; t < 2; ++t) {
      v16bf vb = frag_from_bf16(vts + (t * 16) * 64 + k0, 64);
      oacc[t] = WMMA_BF16(ap, vb, oacc[t]);
    }
  }
#pragma unroll
  for (int t = 0; t < 2; ++t)
#pragma unroll
    for (int j = 0; j < 8; ++j) {
      const int n = wv * 16 + j + 8 * half;
      if (n < NSEQ)
        attout[((size_t)b * NSEQ + n) * 384 + h * 32 + t * 16 + lr] = oacc[t][j];
    }
}

// Fused image-to-text cross attention: keys are exactly 64 text tokens.
// kvtext layout: (32, 64, 768), column = s*384 + h*32 + d.
__global__ __launch_bounds__(128)
void cross_attn_kernel(const float* __restrict__ q_i2t, const float* __restrict__ kvtext,
                       const float* __restrict__ ymask, float* __restrict__ yatt) {
  const int b   = blockIdx.x / HEADS;
  const int h   = blockIdx.x % HEADS;
  const int bt  = b >> 6;  // repeat over nW=64 windows
  const int tid = threadIdx.x;
  const int wv = tid >> 5, lane = tid & 31, half = lane >> 4, lr = lane & 15;

  __shared__ float  qf[64 * 32];
  __shared__ float  kf[64 * 32];
  __shared__ __bf16 vts[32 * 64];
  __shared__ __bf16 ps[64 * 64];
  __shared__ float  sc[64 * 64];

  const float scale = 0.1767766952966369f;

  for (int i = tid + NSEQ * 32; i < 64 * 32; i += 128) qf[i] = 0.f;
#ifdef USE_ASYNC_LDS
  for (int c = tid; c < NSEQ * 8; c += 128) {
    const int n = c >> 3, f4 = (c & 7) << 2;
    async_copy_b128(q_i2t + ((size_t)b * NSEQ + n) * 384 + h * 32 + f4, qf + n * 32 + f4);
  }
  for (int c = tid; c < 64 * 8; c += 128) {
    const int n = c >> 3, f4 = (c & 7) << 2;
    async_copy_b128(kvtext + ((size_t)bt * 64 + n) * 768 + h * 32 + f4, kf + n * 32 + f4);
  }
#else
  for (int i = tid; i < NSEQ * 32; i += 128) {
    const int n = i >> 5, d = i & 31;
    qf[i] = q_i2t[((size_t)b * NSEQ + n) * 384 + h * 32 + d];
  }
  for (int i = tid; i < 64 * 32; i += 128) {
    const int n = i >> 5, d = i & 31;
    kf[i] = kvtext[((size_t)bt * 64 + n) * 768 + h * 32 + d];
  }
#endif
  for (int i = tid; i < 32 * 64; i += 128) {
    const int d = i >> 6, m = i & 63;
    vts[i] = (__bf16)kvtext[((size_t)bt * 64 + m) * 768 + 384 + h * 32 + d];
  }
#ifdef USE_ASYNC_LDS
  __builtin_amdgcn_s_wait_asynccnt(0);
#endif
  __syncthreads();

  {
    v16bf a = frag_from_f32(qf + (wv * 16) * 32, 32);
#pragma unroll
    for (int t = 0; t < 4; ++t) {
      v16bf kb = frag_from_f32(kf + (t * 16) * 32, 32);
      v8f s = {};
      s = WMMA_BF16(a, kb, s);
#pragma unroll
      for (int j = 0; j < 8; ++j) {
        const int row = wv * 16 + j + 8 * half;
        const int col = t * 16 + lr;
        sc[row * 64 + col] = s[j] * scale + ymask[bt * 64 + col];
      }
    }
  }
  __syncthreads();

  if (tid < 64) {
    const int row = tid;
    if (row < NSEQ) {
      float mx = -1e30f;
      for (int m = 0; m < 64; ++m) mx = fmaxf(mx, sc[row * 64 + m]);
      float sum = 0.f;
      for (int m = 0; m < 64; ++m) {
        const float e = __expf(sc[row * 64 + m] - mx);
        sc[row * 64 + m] = e;
        sum += e;
      }
      const float inv = 1.f / sum;
      for (int m = 0; m < 64; ++m) ps[row * 64 + m] = (__bf16)(sc[row * 64 + m] * inv);
    } else {
      for (int m = 0; m < 64; ++m) ps[row * 64 + m] = (__bf16)0.f;
    }
  }
  __syncthreads();

  v8f oacc[2] = {};
  for (int k0 = 0; k0 < 64; k0 += 32) {
    v16bf ap = frag_from_bf16(ps + (wv * 16) * 64 + k0, 64);
#pragma unroll
    for (int t = 0; t < 2; ++t) {
      v16bf vb = frag_from_bf16(vts + (t * 16) * 64 + k0, 64);
      oacc[t] = WMMA_BF16(ap, vb, oacc[t]);
    }
  }
#pragma unroll
  for (int t = 0; t < 2; ++t)
#pragma unroll
    for (int j = 0; j < 8; ++j) {
      const int n = wv * 16 + j + 8 * half;
      if (n < NSEQ)
        yatt[((size_t)b * NSEQ + n) * 384 + h * 32 + t * 16 + lr] = oacc[t][j];
    }
}

extern "C" void kernel_launch(void* const* d_in, const int* in_sizes, int n_in,
                              void* d_out, int out_size, void* d_ws, size_t ws_size,
                              hipStream_t stream) {
  const float* x       = (const float*)d_in[0];
  const float* mask    = (const float*)d_in[1];
  const float* y       = (const float*)d_in[2];
  const float* y_mask  = (const float*)d_in[3];
  const float* w_qkv   = (const float*)d_in[4];
  const float* b_qkv   = (const float*)d_in[5];
  const float* rel_tab = (const float*)d_in[6];
  const float* w_proj  = (const float*)d_in[7];
  const float* b_proj  = (const float*)d_in[8];
  const float* w_kvt   = (const float*)d_in[9];
  const float* b_kvt   = (const float*)d_in[10];
  const float* w_qi    = (const float*)d_in[11];
  const float* b_qi    = (const float*)d_in[12];
  const float* w_pi    = (const float*)d_in[13];
  const float* b_pi    = (const float*)d_in[14];
  float* out = (float*)d_out;

  // Workspace layout (floats). q_i2t reuses qkv (dead after win_attn);
  // yatt reuses attout (dead after the proj GEMM).
  float* ws = (float*)d_ws;
  const size_t QKV = 115605504ull;  // 2048*49*1152
  const size_t ACT = 38535168ull;   // 2048*49*384
  float* qkv    = ws;               // also q_i2t
  float* attout = ws + QKV;         // also yatt
  float* outb   = attout + ACT;
  float* kvtext = outb + ACT;

  const int M = 2048 * 49;  // 100352
  auto nblk = [](int M_, int N_) { return ((M_ / 32) * (N_ / 64) + 7) / 8; };

  // 1) qkv projection
  gemm_bf16_kernel<<<nblk(M, 1152), 256, 0, stream>>>(x, w_qkv, b_qkv, nullptr, qkv, M, 1152, 384);
  // 2) fused window self-attention
  win_attn_kernel<<<2048 * HEADS, 128, 0, stream>>>(qkv, rel_tab, mask, attout);
  // 3) output projection -> out
  gemm_bf16_kernel<<<nblk(M, 384), 256, 0, stream>>>(attout, w_proj, b_proj, nullptr, outb, M, 384, 384);
  // 4) text kv projection
  gemm_bf16_kernel<<<nblk(2048, 768), 256, 0, stream>>>(y, w_kvt, b_kvt, nullptr, kvtext, 2048, 768, 768);
  // 5) q_i2t projection (reuses qkv buffer)
  gemm_bf16_kernel<<<nblk(M, 384), 256, 0, stream>>>(outb, w_qi, b_qi, nullptr, qkv, M, 384, 384);
  // 6) fused cross attention (reuses attout buffer)
  cross_attn_kernel<<<2048 * HEADS, 128, 0, stream>>>(qkv, kvtext, y_mask, attout);
  // 7) final projection + residual add, straight to d_out
  gemm_bf16_kernel<<<nblk(M, 384), 256, 0, stream>>>(attout, w_pi, b_pi, outb, out, M, 384, 384);
}
